// MAB_85160611545094
// MI455X (gfx1250) — matrix operand order repaired and checked by reading, not compiled, not run on hardware
//
#include <hip/hip_runtime.h>

typedef __attribute__((ext_vector_type(16))) _Float16 v16h;
typedef __attribute__((ext_vector_type(8)))  _Float16 h8;
typedef __attribute__((ext_vector_type(8)))  float    v8f;

#define WMMA_F16(a, b, c) \
  __builtin_amdgcn_wmma_f32_16x16x32_f16(false, (a), false, (b), (short)0, (c), false, false)

// ---------------------------------------------------------------------------
// Fragment builders (wave32); A and B share contiguous K order k = 16*(lane/16) + s, not the ISA table order.
// A (16x32 f16): lane%16 -> M; 16 contiguous K halves; kb = 0 (lanes<16) / 16 (lanes>=16)
// B (32x16 f16): lane%16 = N column; 16 contiguous K halves; kb = 0 / 16
// C/D (16x16 f32): lane%16 = N, VGPR v -> M = v + 8*(lane>=16)
// ---------------------------------------------------------------------------

// A fragment from fp32 row-major (converted on the fly)
__device__ __forceinline__ v16h a_frag_f32(const float* __restrict__ p0, int ld) {
  const int l = threadIdx.x & 31;
  const float* p = p0 + (size_t)(l & 15) * ld + ((l & 16) ? 16 : 0);
  float4 x0 = *(const float4*)(p);
  float4 x1 = *(const float4*)(p + 4);
  float4 x2 = *(const float4*)(p + 8);
  float4 x3 = *(const float4*)(p + 12);
  v16h a;
  a[0]=(_Float16)x0.x;  a[1]=(_Float16)x0.y;  a[2]=(_Float16)x0.z;  a[3]=(_Float16)x0.w;
  a[4]=(_Float16)x1.x;  a[5]=(_Float16)x1.y;  a[6]=(_Float16)x1.z;  a[7]=(_Float16)x1.w;
  a[8]=(_Float16)x2.x;  a[9]=(_Float16)x2.y;  a[10]=(_Float16)x2.z; a[11]=(_Float16)x2.w;
  a[12]=(_Float16)x3.x; a[13]=(_Float16)x3.y; a[14]=(_Float16)x3.z; a[15]=(_Float16)x3.w;
  return a;
}

// B fragment: B[k][n] = p0[n*ld + k], f16, 16 contiguous K halves (32B aligned)
__device__ __forceinline__ v16h b_frag_h(const _Float16* __restrict__ p0, int ld) {
  const int l = threadIdx.x & 31;
  return *(const v16h*)(p0 + (size_t)(l & 15) * ld + ((l & 16) ? 16 : 0));
}

// ---------------------------------------------------------------------------
// Kernel 0: transpose + f16-convert the 4 weight matrices (one-shot, 4MB).
// WT[w][n][k] = W[k][n].
// ---------------------------------------------------------------------------
__global__ __launch_bounds__(256) void wt_kernel(
    const float* __restrict__ Wq, const float* __restrict__ Wk,
    const float* __restrict__ Wv, const float* __restrict__ Wo,
    _Float16* __restrict__ WT) {
  const int w = blockIdx.y;
  const float* W = (w == 0) ? Wq : (w == 1) ? Wk : (w == 2) ? Wv : Wo;
  _Float16* dst = WT + (size_t)w * 512 * 512;
  const int idx = blockIdx.x * 256 + threadIdx.x;   // grid.x = 1024
  const int k = idx >> 9, n = idx & 511;
  dst[(size_t)n * 512 + k] = (_Float16)W[idx];
}

// ---------------------------------------------------------------------------
// Kernel 1: projections. Each wave computes a 16x64 tile, K = 16 steps of 32.
//  z=0: Qp fp32 (residual + Q A-frags); z=1: Kp f16 row-major;
//  z=2: Vp f16 transposed VpT[b][d][k].
// ---------------------------------------------------------------------------
__global__ __launch_bounds__(256) void proj_kernel(
    const float* __restrict__ Q, const float* __restrict__ K,
    const _Float16* __restrict__ WT,
    const float* __restrict__ bq, const float* __restrict__ bk,
    const float* __restrict__ bv,
    float* __restrict__ Qp, _Float16* __restrict__ Kph,
    _Float16* __restrict__ VpT) {
  const int z = blockIdx.y;
  const float* X = (z == 0) ? Q : K;
  const _Float16* WTz = WT + (size_t)z * 512 * 512;
  const float* bias = (z == 0) ? bq : (z == 1) ? bk : bv;

  const int gw = blockIdx.x * 8 + (threadIdx.x >> 5);
  const int m0 = (gw >> 3) * 16;    // 512 M-tiles
  const int n0 = (gw & 7) * 64;     // 8 N-blocks of 64

  v8f acc[4] = {};
#pragma unroll 2
  for (int kt = 0; kt < 16; ++kt) {
    const int k0 = kt * 32;
    // batch all loads ahead of the WMMA burst
    v16h a   = a_frag_f32(X + (size_t)m0 * 512 + k0, 512);
    v16h bf0 = b_frag_h(WTz + (size_t)(n0 +  0) * 512 + k0, 512);
    v16h bf1 = b_frag_h(WTz + (size_t)(n0 + 16) * 512 + k0, 512);
    v16h bf2 = b_frag_h(WTz + (size_t)(n0 + 32) * 512 + k0, 512);
    v16h bf3 = b_frag_h(WTz + (size_t)(n0 + 48) * 512 + k0, 512);
    acc[0] = WMMA_F16(a, bf0, acc[0]);
    acc[1] = WMMA_F16(a, bf1, acc[1]);
    acc[2] = WMMA_F16(a, bf2, acc[2]);
    acc[3] = WMMA_F16(a, bf3, acc[3]);
  }

  const int l  = threadIdx.x & 31;
  const int n  = l & 15;
  const int mb = (l & 16) ? 8 : 0;
  if (z == 0) {
#pragma unroll
    for (int j = 0; j < 4; ++j) {
      const int col = n0 + j * 16 + n;
      const float bb = bias[col];
#pragma unroll
      for (int v = 0; v < 8; ++v)
        Qp[(size_t)(m0 + mb + v) * 512 + col] = acc[j][v] + bb;
    }
  } else if (z == 1) {
#pragma unroll
    for (int j = 0; j < 4; ++j) {
      const int col = n0 + j * 16 + n;
      const float bb = bias[col];
#pragma unroll
      for (int v = 0; v < 8; ++v)
        Kph[(size_t)(m0 + mb + v) * 512 + col] = (_Float16)(acc[j][v] + bb);
    }
  } else {
    const int b  = m0 >> 11;             // batch (16-row tiles never cross)
    const int kr = (m0 & 2047) + mb;     // k position within batch
#pragma unroll
    for (int j = 0; j < 4; ++j) {
      const int col = n0 + j * 16 + n;
      const float bb = bias[col];
      h8 hh;
#pragma unroll
      for (int v = 0; v < 8; ++v) hh[v] = (_Float16)(acc[j][v] + bb);
      *(h8*)(VpT + ((size_t)(b * 512 + col)) * 2048 + kr) = hh;
    }
  }
}

// ---------------------------------------------------------------------------
// Kernel 2: attention. One workgroup (4 waves) per (h, b, 16-row q-tile).
// 128KB fp32 logits stripe -> exact softmax -> A written once -> AV from LDS.
// ~129KB LDS so two workgroups co-reside per 320KB WGP (latency hiding).
// ---------------------------------------------------------------------------
__global__ __launch_bounds__(128) void attn_kernel(
    const float* __restrict__ Qp, const _Float16* __restrict__ Kph,
    const _Float16* __restrict__ VpT,
    float* __restrict__ Oh, float* __restrict__ Aout) {
  __shared__ float S[16][2048];      // 128 KB logits -> probabilities
  __shared__ float red[16][8];
  __shared__ float rowmax[16], rowinv[16];

  const int q0 = blockIdx.x * 16;
  const int b  = blockIdx.y;
  const int h  = blockIdx.z;

  const int wave = threadIdx.x >> 5;
  const int l    = threadIdx.x & 31;
  const float scale = 0.044194173824159216f;  // 1/sqrt(512)

  // ---- Stage 1: logits. Wave w covers S columns [w*512, w*512+512).
  const float* Qbase = Qp + ((size_t)b * 2048 + q0) * 512 + h * 64;
  v16h aq0 = a_frag_f32(Qbase + 0, 512);
  v16h aq1 = a_frag_f32(Qbase + 32, 512);
#pragma unroll 2
  for (int t = 0; t < 32; ++t) {
    const int kcol = wave * 512 + t * 16;
    const _Float16* Kbase = Kph + ((size_t)b * 2048 + kcol) * 512 + h * 64;
    v16h bf0 = b_frag_h(Kbase + 0, 512);
    v16h bf1 = b_frag_h(Kbase + 32, 512);
    v8f c = {};
    c = WMMA_F16(aq0, bf0, c);
    c = WMMA_F16(aq1, bf1, c);
    const int n  = l & 15;
    const int mb = (l & 16) ? 8 : 0;
#pragma unroll
    for (int v = 0; v < 8; ++v) S[mb + v][kcol + n] = c[v] * scale;
  }
  __syncthreads();

  // ---- Stage 2: softmax. 8 threads per row, 256-col chunks (float4).
  const int r  = threadIdx.x >> 3;
  const int s8 = threadIdx.x & 7;
  const int c0 = s8 * 256;
  float mx = -3.4e38f;
#pragma unroll 4
  for (int c4 = 0; c4 < 64; ++c4) {
    float4 x = *(const float4*)&S[r][c0 + c4 * 4];
    mx = fmaxf(mx, fmaxf(fmaxf(x.x, x.y), fmaxf(x.z, x.w)));
  }
  red[r][s8] = mx;
  __syncthreads();
  if (s8 == 0) {
    float m = red[r][0];
    for (int i = 1; i < 8; ++i) m = fmaxf(m, red[r][i]);
    rowmax[r] = m;
  }
  __syncthreads();
  const float rm = rowmax[r];
  float sum = 0.f;
#pragma unroll 4
  for (int c4 = 0; c4 < 64; ++c4) {
    float4 x = *(const float4*)&S[r][c0 + c4 * 4];
    x.x = __expf(x.x - rm); x.y = __expf(x.y - rm);
    x.z = __expf(x.z - rm); x.w = __expf(x.w - rm);
    *(float4*)&S[r][c0 + c4 * 4] = x;
    sum += (x.x + x.y) + (x.z + x.w);
  }
  red[r][s8] = sum;
  __syncthreads();
  if (s8 == 0) {
    float s = 0.f;
    for (int i = 0; i < 8; ++i) s += red[r][i];
    rowinv[r] = 1.f / s;
  }
  __syncthreads();

  // ---- Stage 3: normalize in LDS + single coalesced fp32 write of A.
  float* Arow = Aout + ((size_t)(h * 4 + b) * 2048 + q0) * 2048;
  for (int it = 0; it < 64; ++it) {
    int idx4 = it * 128 + threadIdx.x;  // 8192 float4s
    int rr = idx4 >> 9;
    int cc = (idx4 & 511) * 4;
    float4 vv = *(const float4*)&S[rr][cc];
    const float inv = rowinv[rr];
    vv.x *= inv; vv.y *= inv; vv.z *= inv; vv.w *= inv;
    *(float4*)&S[rr][cc] = vv;
    *(float4*)(Arow + (size_t)rr * 2048 + cc) = vv;
  }
  __syncthreads();

  // ---- Stage 4: Oh = Qh + A @ Vh. Wave w owns head cols [w*16, w*16+16).
  // Manual unroll x4; all loads batched ahead of the WMMA group; two
  // accumulator chains break the C-operand dependency.
  const int ncol = h * 64 + wave * 16;
  const _Float16* Vb = VpT + ((size_t)b * 512 + ncol) * 2048;
  v8f cacc0 = {}, cacc1 = {};
  for (int kt = 0; kt < 64; kt += 4) {
    const int k0 = kt * 32;
    v16h a0 = a_frag_f32(&S[0][0] + k0 +  0, 2048);
    v16h a1 = a_frag_f32(&S[0][0] + k0 + 32, 2048);
    v16h a2 = a_frag_f32(&S[0][0] + k0 + 64, 2048);
    v16h a3 = a_frag_f32(&S[0][0] + k0 + 96, 2048);
    v16h b0f = b_frag_h(Vb + k0 +  0, 2048);
    v16h b1f = b_frag_h(Vb + k0 + 32, 2048);
    v16h b2f = b_frag_h(Vb + k0 + 64, 2048);
    v16h b3f = b_frag_h(Vb + k0 + 96, 2048);
    cacc0 = WMMA_F16(a0, b0f, cacc0);
    cacc1 = WMMA_F16(a1, b1f, cacc1);
    cacc0 = WMMA_F16(a2, b2f, cacc0);
    cacc1 = WMMA_F16(a3, b3f, cacc1);
  }
  const int n  = l & 15;
  const int mb = (l & 16) ? 8 : 0;
#pragma unroll
  for (int v = 0; v < 8; ++v) {
    const size_t ro = ((size_t)b * 2048 + q0 + mb + v) * 512 + ncol + n;
    Oh[ro] = (cacc0[v] + cacc1[v]) + Qp[ro];
  }
}

// ---------------------------------------------------------------------------
// Kernel 3: O = LN1( X + relu(X @ Wo + bo) ), X = LN0(Oh). 16 rows per block.
// ---------------------------------------------------------------------------
__global__ __launch_bounds__(256) void post_kernel(
    const float* __restrict__ Oh, const _Float16* __restrict__ WoT,
    const float* __restrict__ bo,
    const float* __restrict__ g0, const float* __restrict__ b0,
    const float* __restrict__ g1, const float* __restrict__ b1,
    float* __restrict__ Out) {
  __shared__ float X[16][512];
  __shared__ float Y[16][512];
  __shared__ float red1[16][16];
  __shared__ float red2[16][16];
  __shared__ float rmean[16], rrstd[16];

  const int row0 = blockIdx.x * 16;

  for (int it = 0; it < 8; ++it) {
    int idx4 = it * 256 + threadIdx.x;   // 2048 float4s
    int rr = idx4 >> 7;
    int cc = (idx4 & 127) * 4;
    *(float4*)&X[rr][cc] = *(const float4*)(Oh + (size_t)(row0 + rr) * 512 + cc);
  }
  __syncthreads();

  const int r  = threadIdx.x >> 4;
  const int s  = threadIdx.x & 15;
  const int cb = s * 32;

  // LN0 (float4 LDS passes)
  {
    float sum = 0.f, sumsq = 0.f;
#pragma unroll
    for (int c4 = 0; c4 < 8; ++c4) {
      float4 x = *(const float4*)&X[r][cb + c4 * 4];
      sum   += (x.x + x.y) + (x.z + x.w);
      sumsq += (x.x * x.x + x.y * x.y) + (x.z * x.z + x.w * x.w);
    }
    red1[r][s] = sum; red2[r][s] = sumsq;
    __syncthreads();
    if (s == 0) {
      float S1 = 0.f, S2 = 0.f;
      for (int i = 0; i < 16; ++i) { S1 += red1[r][i]; S2 += red2[r][i]; }
      float m = S1 * (1.f / 512.f);
      rmean[r] = m;
      rrstd[r] = rsqrtf(S2 * (1.f / 512.f) - m * m + 1e-5f);
    }
    __syncthreads();
    const float m = rmean[r], rs = rrstd[r];
#pragma unroll
    for (int c4 = 0; c4 < 8; ++c4) {
      const int c = cb + c4 * 4;
      float4 x = *(const float4*)&X[r][c];
      float4 g = *(const float4*)&g0[c];
      float4 bb = *(const float4*)&b0[c];
      x.x = (x.x - m) * rs * g.x + bb.x;
      x.y = (x.y - m) * rs * g.y + bb.y;
      x.z = (x.z - m) * rs * g.z + bb.z;
      x.w = (x.w - m) * rs * g.w + bb.w;
      *(float4*)&X[r][c] = x;
    }
    __syncthreads();
  }

  // GEMM: Y = X @ Wo ; wave w owns cols [w*64, w*64+64)
  const int wave = threadIdx.x >> 5;
  const int l    = threadIdx.x & 31;
  v8f acc[4] = {};
#pragma unroll 2
  for (int kt = 0; kt < 16; ++kt) {
    const int k0 = kt * 32;
    v16h a   = a_frag_f32(&X[0][0] + k0, 512);
    v16h bf0 = b_frag_h(WoT + (size_t)(wave * 64 +  0) * 512 + k0, 512);
    v16h bf1 = b_frag_h(WoT + (size_t)(wave * 64 + 16) * 512 + k0, 512);
    v16h bf2 = b_frag_h(WoT + (size_t)(wave * 64 + 32) * 512 + k0, 512);
    v16h bf3 = b_frag_h(WoT + (size_t)(wave * 64 + 48) * 512 + k0, 512);
    acc[0] = WMMA_F16(a, bf0, acc[0]);
    acc[1] = WMMA_F16(a, bf1, acc[1]);
    acc[2] = WMMA_F16(a, bf2, acc[2]);
    acc[3] = WMMA_F16(a, bf3, acc[3]);
  }
  const int n  = l & 15;
  const int mb = (l & 16) ? 8 : 0;
#pragma unroll
  for (int j = 0; j < 4; ++j) {
    const int col = wave * 64 + j * 16 + n;
    const float bb = bo[col];
#pragma unroll
    for (int v = 0; v < 8; ++v) {
      const int m = mb + v;
      Y[m][col] = X[m][col] + fmaxf(acc[j][v] + bb, 0.f);
    }
  }
  __syncthreads();

  // LN1 -> Out (float4 passes)
  {
    float sum = 0.f, sumsq = 0.f;
#pragma unroll
    for (int c4 = 0; c4 < 8; ++c4) {
      float4 x = *(const float4*)&Y[r][cb + c4 * 4];
      sum   += (x.x + x.y) + (x.z + x.w);
      sumsq += (x.x * x.x + x.y * x.y) + (x.z * x.z + x.w * x.w);
    }
    red1[r][s] = sum; red2[r][s] = sumsq;
    __syncthreads();
    if (s == 0) {
      float S1 = 0.f, S2 = 0.f;
      for (int i = 0; i < 16; ++i) { S1 += red1[r][i]; S2 += red2[r][i]; }
      float m = S1 * (1.f / 512.f);
      rmean[r] = m;
      rrstd[r] = rsqrtf(S2 * (1.f / 512.f) - m * m + 1e-5f);
    }
    __syncthreads();
    const float m = rmean[r], rs = rrstd[r];
#pragma unroll
    for (int c4 = 0; c4 < 8; ++c4) {
      const int c = cb + c4 * 4;
      float4 x = *(const float4*)&Y[r][c];
      float4 g = *(const float4*)&g1[c];
      float4 bb = *(const float4*)&b1[c];
      x.x = (x.x - m) * rs * g.x + bb.x;
      x.y = (x.y - m) * rs * g.y + bb.y;
      x.z = (x.z - m) * rs * g.z + bb.z;
      x.w = (x.w - m) * rs * g.w + bb.w;
      *(float4*)(Out + (size_t)(row0 + r) * 512 + c) = x;
    }
  }
}

// ---------------------------------------------------------------------------
extern "C" void kernel_launch(void* const* d_in, const int* in_sizes, int n_in,
                              void* d_out, int out_size, void* d_ws, size_t ws_size,
                              hipStream_t stream) {
  const float* Q  = (const float*)d_in[0];
  const float* K  = (const float*)d_in[1];
  const float* Wq = (const float*)d_in[2];
  const float* bq = (const float*)d_in[3];
  const float* Wk = (const float*)d_in[4];
  const float* bk = (const float*)d_in[5];
  const float* Wv = (const float*)d_in[6];
  const float* bv = (const float*)d_in[7];
  const float* Wo = (const float*)d_in[8];
  const float* bo = (const float*)d_in[9];
  const float* g0 = (const float*)d_in[10];
  const float* b0 = (const float*)d_in[11];
  const float* g1 = (const float*)d_in[12];
  const float* b1 = (const float*)d_in[13];

  float* out  = (float*)d_out;
  float* Aout = out + 4ull * 2048 * 512;             // O first, then A

  char* wsb = (char*)d_ws;
  float*    Qp  = (float*)wsb;                       // 16 MB fp32
  float*    Oh  = (float*)(wsb + (16ull << 20));     // 16 MB fp32
  _Float16* Kph = (_Float16*)(wsb + (32ull << 20));  //  8 MB f16
  _Float16* VpT = (_Float16*)(wsb + (40ull << 20));  //  8 MB f16 (transposed)
  _Float16* WT  = (_Float16*)(wsb + (48ull << 20));  //  2 MB f16 (4 weights^T)

  wt_kernel  <<<dim3(1024, 4), 256, 0, stream>>>(Wq, Wk, Wv, Wo, WT);
  proj_kernel<<<dim3(512, 3),  256, 0, stream>>>(Q, K, WT, bq, bk, bv, Qp, Kph, VpT);
  attn_kernel<<<dim3(128, 4, 8), 128, 0, stream>>>(Qp, Kph, VpT, Oh, Aout);
  post_kernel<<<dim3(512), 256, 0, stream>>>(Oh, WT + 3ull * 512 * 512, bo,
                                             g0, b0, g1, b1, out);
}
